// BinaryMaskingModule_5549097746947
// MI455X (gfx1250) — matrix operand order repaired and checked by reading, not compiled, not run on hardware
//
#include <hip/hip_runtime.h>
#include <hip/hip_bf16.h>

#define B_SZ  8
#define N_PTS 4096
#define C_DIM 256

typedef __attribute__((ext_vector_type(16))) __bf16 v16bf;
typedef __attribute__((ext_vector_type(8)))  float  v8f;

union FragBF {
  v16bf bf;
  unsigned short us[16];
  uint4 q[2];
};

__device__ __forceinline__ unsigned short f2bf(float x) {
  unsigned int u = __float_as_uint(x);
  u += 0x7FFFu + ((u >> 16) & 1u);   // round-to-nearest-even
  return (unsigned short)(u >> 16);
}

__device__ __forceinline__ v8f v8f_zero() {
  v8f z = {0.f, 0.f, 0.f, 0.f, 0.f, 0.f, 0.f, 0.f};
  return z;
}

// A-matrix (16x32 bf16, MxK): lane m=lane&15; lanes<16 hold K {0..7,16..23},
// lanes>=16 hold K {8..15,24..31}. rowk = &A[m][kbase] (row-major, 16B aligned).
__device__ __forceinline__ v16bf load_a_frag(const unsigned short* rowk, int lane) {
  FragBF f;
  const unsigned short* p = rowk + ((lane & 16) ? 8 : 0);
  f.q[0] = *reinterpret_cast<const uint4*>(p);
  f.q[1] = *reinterpret_cast<const uint4*>(p + 16);
  return f.bf;
}

// B-matrix (32x16 bf16, KxN): lane holds column n=lane&15; lanes<16 hold K {0..7,16..23},
// lanes>=16 hold K {8..15,24..31}. rowk = &Bcol_row[kbase].
__device__ __forceinline__ v16bf load_b_frag(const unsigned short* rowk, int lane) {
  FragBF f;
  const unsigned short* p = rowk + ((lane & 16) ? 8 : 0);
  f.q[0] = *reinterpret_cast<const uint4*>(p);
  f.q[1] = *reinterpret_cast<const uint4*>(p + 16);
  return f.bf;
}

__device__ __forceinline__ v8f wmma_bf16(v16bf a, v16bf b, v8f c) {
  return __builtin_amdgcn_wmma_f32_16x16x32_bf16(false, a, false, b, (short)0, c, false, false);
}

// ---------------------------------------------------------------- utilities
__global__ void k_cvt_bf16(const float* __restrict__ src,
                           unsigned short* __restrict__ dst, int n) {
  int i = blockIdx.x * blockDim.x + threadIdx.x;
  if (i < n) dst[i] = f2bf(src[i]);
}

__global__ void k_rownorm(const float* __restrict__ x,
                          unsigned short* __restrict__ xbf,
                          float* __restrict__ xx, int total) {
  int p = blockIdx.x * blockDim.x + threadIdx.x;
  if (p >= total) return;
  const float* r = x + (size_t)p * 64;
  unsigned short* o = xbf + (size_t)p * 64;
  float s = 0.f;
#pragma unroll
  for (int c = 0; c < 64; ++c) { float v = r[c]; s += v * v; o[c] = f2bf(v); }
  xx[p] = s;
}

// ------------------------------------------- pointwise conv: Y=relu(X*W^T+b)
template <int K, int O>
__global__ __launch_bounds__(256)
void k_pw_relu(const float* __restrict__ X, const unsigned short* __restrict__ W,
               const float* __restrict__ bias, float* __restrict__ Y) {
  __shared__ unsigned short XS[64 * K];
  const int tid = threadIdx.x, lane = tid & 31, wave = tid >> 5;
  const int m0 = blockIdx.x * 64;
  for (int e = tid; e < 64 * K; e += 256) XS[e] = f2bf(X[(size_t)m0 * K + e]);
  __syncthreads();
  constexpr int NT = O / 16;
  for (int t = wave; t < 4 * NT; t += 8) {
    const int mt = t / NT, nt = t % NT;
    v8f acc = v8f_zero();
    const unsigned short* arow = &XS[(mt * 16 + (lane & 15)) * K];
    const unsigned short* brow = W + (size_t)(nt * 16 + (lane & 15)) * K;
#pragma unroll
    for (int kb = 0; kb < K; kb += 32)
      acc = wmma_bf16(load_a_frag(arow + kb, lane), load_b_frag(brow + kb, lane), acc);
    const int col = nt * 16 + (lane & 15);
    const float bb = bias[col];
#pragma unroll
    for (int r = 0; r < 8; ++r) {
      int row = mt * 16 + r + ((lane & 16) ? 8 : 0);
      float v = acc[r] + bb;
      Y[(size_t)(m0 + row) * O + col] = v > 0.f ? v : 0.f;
    }
  }
}

// ---------------------------------------------- fused kNN (distances + top-k)
template <int KN>
__global__ __launch_bounds__(128)
void k_knn(const unsigned short* __restrict__ xbf, const float* __restrict__ xx,
           int* __restrict__ idxO) {
  __shared__ float dtile[4][16][17];
  __shared__ float md[4][16][KN];
  __shared__ int   mi[4][16][KN];
  const int tid = threadIdx.x, lane = tid & 31, wave = tid >> 5;
  const int b = blockIdx.y;
  const int rowbase = blockIdx.x * 16;
  const size_t xoff = (size_t)b * N_PTS;

  const unsigned short* arow = xbf + (xoff + rowbase + (lane & 15)) * 64;
  v16bf a0 = load_a_frag(arow + 0, lane);
  v16bf a1 = load_a_frag(arow + 32, lane);
  float xr[8];
#pragma unroll
  for (int r = 0; r < 8; ++r)
    xr[r] = xx[xoff + rowbase + r + ((lane & 16) ? 8 : 0)];

  float bestd[KN]; int besti[KN];
#pragma unroll
  for (int s = 0; s < KN; ++s) { bestd[s] = 3.0e38f; besti[s] = 0; }

  const int colstart = wave * 1024;
  for (int ct = 0; ct < 64; ++ct) {
    const int colbase = colstart + ct * 16;
    const unsigned short* brow = xbf + (xoff + colbase + (lane & 15)) * 64;
    v8f acc = v8f_zero();
    acc = wmma_bf16(a0, load_b_frag(brow, lane), acc);
    acc = wmma_bf16(a1, load_b_frag(brow + 32, lane), acc);
    const float xc = xx[xoff + colbase + (lane & 15)];
#pragma unroll
    for (int r = 0; r < 8; ++r) {
      int rowl = r + ((lane & 16) ? 8 : 0);
      dtile[wave][rowl][lane & 15] = xr[r] + xc - 2.0f * acc[r];
    }
    __syncthreads();
    if (lane < 16) {
#pragma unroll
      for (int c = 0; c < 16; ++c) {
        float d = dtile[wave][lane][c];
        int ci = colbase + c;
        if (d < bestd[KN - 1]) {          // unrolled insertion: stays in VGPRs
          bestd[KN - 1] = d; besti[KN - 1] = ci;
#pragma unroll
          for (int s = KN - 1; s > 0; --s) {
            if (bestd[s] < bestd[s - 1]) {
              float td = bestd[s]; bestd[s] = bestd[s - 1]; bestd[s - 1] = td;
              int ti = besti[s]; besti[s] = besti[s - 1]; besti[s - 1] = ti;
            }
          }
        }
      }
    }
    __syncthreads();
  }
  if (lane < 16) {
#pragma unroll
    for (int s = 0; s < KN; ++s) { md[wave][lane][s] = bestd[s]; mi[wave][lane][s] = besti[s]; }
  }
  __syncthreads();
  if (wave == 0 && lane < 16) {   // 4-way sorted-list merge
    int ptr0 = 0, ptr1 = 0, ptr2 = 0, ptr3 = 0;
    int* orow = idxO + (xoff + rowbase + lane) * KN;
    for (int s = 0; s < KN; ++s) {
      float v0 = (ptr0 < KN) ? md[0][lane][ptr0] : 3.9e38f;
      float v1 = (ptr1 < KN) ? md[1][lane][ptr1] : 3.9e38f;
      float v2 = (ptr2 < KN) ? md[2][lane][ptr2] : 3.9e38f;
      float v3 = (ptr3 < KN) ? md[3][lane][ptr3] : 3.9e38f;
      float best = v0; int bq = 0;
      if (v1 < best) { best = v1; bq = 1; }
      if (v2 < best) { best = v2; bq = 2; }
      if (v3 < best) { best = v3; bq = 3; }
      int pick;
      if (bq == 0)      { pick = mi[0][lane][ptr0]; ptr0++; }
      else if (bq == 1) { pick = mi[1][lane][ptr1]; ptr1++; }
      else if (bq == 2) { pick = mi[2][lane][ptr2]; ptr2++; }
      else              { pick = mi[3][lane][ptr3]; ptr3++; }
      orow[s] = pick;
    }
  }
}

// ------------------------- fused EdgeConv: gather + MLP(s) + k-aggregation
// SECOND=true: two MLPs, max-aggregate (block 0, k=12).
// SECOND=false: one MLP, sum-aggregate (block 1, k=8).
template <int KN, bool SECOND>
__global__ __launch_bounds__(256)
void k_edgeconv(const float* __restrict__ x, const int* __restrict__ idxG,
                const unsigned short* __restrict__ W1, const float* __restrict__ b1,
                const unsigned short* __restrict__ W2, const float* __restrict__ b2,
                float* __restrict__ out) {
  constexpr int P  = (KN == 12) ? 4 : (64 / KN);  // points per workgroup
  constexpr int RV = P * KN;                       // valid edge rows (<=64)
  __shared__ unsigned short E[64 * 128];
  __shared__ unsigned short H[SECOND ? 64 * 256 : 1];
  __shared__ float aggS[P * 256];
  __shared__ int nbIdx[64];
  const int tid = threadIdx.x, lane = tid & 31, wave = tid >> 5;
  const int b = blockIdx.y;
  const int p0 = blockIdx.x * P;
  const size_t xbase = (size_t)b * N_PTS;

  if (tid < RV) nbIdx[tid] = idxG[(xbase + p0 + tid / KN) * KN + tid % KN];
  for (int e = tid; e < P * 256; e += 256) aggS[e] = 0.0f;
  __syncthreads();

  // build edge features [xj - xi, xi] as bf16 in LDS
  for (int e = tid; e < 64 * 128; e += 256) {
    int r = e >> 7, c = e & 127;
    float v = 0.f;
    if (r < RV) {
      int p = r / KN;
      int n = p0 + p;
      int nb = nbIdx[r];
      const float* xi = x + (xbase + n) * 64;
      const float* xj = x + (xbase + nb) * 64;
      v = (c < 64) ? (xj[c] - xi[c]) : xi[c - 64];
    }
    E[e] = f2bf(v);
  }
  __syncthreads();

  // MLP1: [64x128] @ W1[256x128]^T
  for (int t = wave; t < 64; t += 8) {
    int mt = t & 3, nt = t >> 2;
    v8f acc = v8f_zero();
    const unsigned short* arow = &E[(mt * 16 + (lane & 15)) * 128];
    const unsigned short* brow = W1 + (size_t)(nt * 16 + (lane & 15)) * 128;
#pragma unroll
    for (int kb = 0; kb < 128; kb += 32)
      acc = wmma_bf16(load_a_frag(arow + kb, lane), load_b_frag(brow + kb, lane), acc);
    int col = nt * 16 + (lane & 15);
    float bb = b1[col];
#pragma unroll
    for (int r = 0; r < 8; ++r) {
      int row = mt * 16 + r + ((lane & 16) ? 8 : 0);
      float v = acc[r] + bb;
      v = v > 0.f ? v : 0.f;
      if constexpr (SECOND) {
        H[row * 256 + col] = f2bf(v);
      } else {
        if (row < RV) atomicAdd(&aggS[(row / KN) * 256 + col], v);  // sum-agg
      }
    }
  }
  __syncthreads();

  if constexpr (SECOND) {
    // MLP2: [64x256] @ W2[256x256]^T, then max-aggregate (values >= 0 post-ReLU)
    for (int t = wave; t < 64; t += 8) {
      int mt = t & 3, nt = t >> 2;
      v8f acc = v8f_zero();
      const unsigned short* arow = &H[(mt * 16 + (lane & 15)) * 256];
      const unsigned short* brow = W2 + (size_t)(nt * 16 + (lane & 15)) * 256;
#pragma unroll
      for (int kb = 0; kb < 256; kb += 32)
        acc = wmma_bf16(load_a_frag(arow + kb, lane), load_b_frag(brow + kb, lane), acc);
      int col = nt * 16 + (lane & 15);
      float bb = b2[col];
#pragma unroll
      for (int r = 0; r < 8; ++r) {
        int row = mt * 16 + r + ((lane & 16) ? 8 : 0);
        float v = acc[r] + bb;
        v = v > 0.f ? v : 0.f;
        if (row < RV)
          atomicMax((int*)&aggS[(row / KN) * 256 + col], __float_as_int(v));
      }
    }
    __syncthreads();
  }

  for (int e = tid; e < P * 256; e += 256) {
    int p = e >> 8, col = e & 255;
    out[(xbase + p0 + p) * 256 + col] = aggS[e];
  }
}

// ------------------------------------------------------------- final 1x1->1
__global__ void k_out(const float* __restrict__ X, const float* __restrict__ w,
                      const float* __restrict__ bias, float* __restrict__ y, int total) {
  int p = blockIdx.x * blockDim.x + threadIdx.x;
  if (p >= total) return;
  const float* r = X + (size_t)p * 64;
  float s = bias[0];
#pragma unroll
  for (int c = 0; c < 64; ++c) s += r[c] * w[c];
  y[p] = s > 0.f ? s : 0.f;
}

extern "C" void kernel_launch(void* const* d_in, const int* in_sizes, int n_in,
                              void* d_out, int out_size, void* d_ws, size_t ws_size,
                              hipStream_t stream) {
  (void)in_sizes; (void)n_in; (void)out_size; (void)ws_size;
  const float* feature = (const float*)d_in[0];
  const float* conv0_w = (const float*)d_in[1];
  const float* conv0_b = (const float*)d_in[2];
  const float* ec1_w1  = (const float*)d_in[3];
  const float* ec1_b1  = (const float*)d_in[4];
  const float* ec1_w2  = (const float*)d_in[5];
  const float* ec1_b2  = (const float*)d_in[6];
  const float* conv1_w = (const float*)d_in[7];
  const float* conv1_b = (const float*)d_in[8];
  const float* ec2_w   = (const float*)d_in[9];
  const float* ec2_b   = (const float*)d_in[10];
  const float* dec1_w  = (const float*)d_in[11];
  const float* dec1_b  = (const float*)d_in[12];
  const float* dec2_w  = (const float*)d_in[13];
  const float* dec2_b  = (const float*)d_in[14];
  const float* out_w   = (const float*)d_in[15];
  const float* out_b   = (const float*)d_in[16];

  const int BN = B_SZ * N_PTS;  // 32768
  char* base = (char*)d_ws;
  size_t off = 0;
  auto alloc = [&](size_t bytes) -> char* {
    char* p = base + off;
    off += (bytes + 255) & ~(size_t)255;
    return p;
  };

  float*          A       = (float*)alloc((size_t)BN * 256 * 4);  // activations [B,N,256]
  float*          x_small = (float*)alloc((size_t)BN * 64 * 4);   // [B,N,64]
  float*          Bf      = (float*)alloc((size_t)BN * 128 * 4);  // decoder mid
  unsigned short* xbf     = (unsigned short*)alloc((size_t)BN * 64 * 2);
  float*          xx      = (float*)alloc((size_t)BN * 4);
  int*            idxb    = (int*)alloc((size_t)BN * 12 * 4);
  unsigned short* wbf0 = (unsigned short*)alloc(64 * 256 * 2);
  unsigned short* wbf1 = (unsigned short*)alloc(256 * 128 * 2);
  unsigned short* wbf2 = (unsigned short*)alloc(256 * 256 * 2);
  unsigned short* wbf3 = (unsigned short*)alloc(64 * 256 * 2);
  unsigned short* wbf4 = (unsigned short*)alloc(256 * 128 * 2);
  unsigned short* wbf5 = (unsigned short*)alloc(128 * 256 * 2);
  unsigned short* wbf6 = (unsigned short*)alloc(64 * 128 * 2);

  auto cvt = [&](const float* s, unsigned short* d, int n) {
    k_cvt_bf16<<<(n + 255) / 256, 256, 0, stream>>>(s, d, n);
  };
  cvt(conv0_w, wbf0, 64 * 256);
  cvt(ec1_w1,  wbf1, 256 * 128);
  cvt(ec1_w2,  wbf2, 256 * 256);
  cvt(conv1_w, wbf3, 64 * 256);
  cvt(ec2_w,   wbf4, 256 * 128);
  cvt(dec1_w,  wbf5, 128 * 256);
  cvt(dec2_w,  wbf6, 64 * 128);

  // block 0
  k_pw_relu<256, 64><<<BN / 64, 256, 0, stream>>>(feature, wbf0, conv0_b, x_small);
  k_rownorm<<<BN / 256, 256, 0, stream>>>(x_small, xbf, xx, BN);
  k_knn<12><<<dim3(N_PTS / 16, B_SZ), 128, 0, stream>>>(xbf, xx, idxb);
  k_edgeconv<12, true><<<dim3(N_PTS / 4, B_SZ), 256, 0, stream>>>(
      x_small, idxb, wbf1, ec1_b1, wbf2, ec1_b2, A);
  // block 1
  k_pw_relu<256, 64><<<BN / 64, 256, 0, stream>>>(A, wbf3, conv1_b, x_small);
  k_rownorm<<<BN / 256, 256, 0, stream>>>(x_small, xbf, xx, BN);
  k_knn<8><<<dim3(N_PTS / 16, B_SZ), 128, 0, stream>>>(xbf, xx, idxb);
  k_edgeconv<8, false><<<dim3(N_PTS / 8, B_SZ), 256, 0, stream>>>(
      x_small, idxb, wbf4, ec2_b, nullptr, nullptr, A);
  // decoder
  k_pw_relu<256, 128><<<BN / 64, 256, 0, stream>>>(A, wbf5, dec1_b, Bf);
  k_pw_relu<128, 64><<<BN / 64, 256, 0, stream>>>(Bf, wbf6, dec2_b, x_small);
  k_out<<<BN / 256, 256, 0, stream>>>(x_small, out_w, out_b, (float*)d_out, BN);
}
